// OctreeConvK2_21337397526842
// MI455X (gfx1250) — matrix operand order repaired and checked by reading, not compiled, not run on hardware
//
#include <hip/hip_runtime.h>

typedef __attribute__((ext_vector_type(16))) __bf16        v16bf;
typedef __attribute__((ext_vector_type(8)))  float         v8f;
typedef __attribute__((ext_vector_type(8)))  unsigned int  v8u;

#define C_IN      128
#define C_OUT     256
#define GEMM_M    131072      // N/8 parent rows with data
#define GEMM_K    1024        // 8*C_IN
#define NPARENT   262144
#define BM        128
#define BN        128
#define BK        32
#define NK        (GEMM_K / BK)   // 32

// LDS layout (bytes)
#define A_STRIDE  80          // 32 bf16 + pad; 16B aligned, conflict-free frag reads
#define A_BUF     (BM * A_STRIDE)          // 10240
#define B_STRIDE  272         // 128 bf16 + 16B pad; 16B aligned
#define B_BUF     (BK * B_STRIDE)          // 8704

// fp32 x2 -> packed bf16x2: single v_cvt_pk_bf16_f32 (RNE) via inline asm
__device__ __forceinline__ unsigned int pack_bf16x2(float a, float b) {
  unsigned int r;
  asm("v_cvt_pk_bf16_f32 %0, %1, %2" : "=v"(r) : "v"(a), "v"(b));
  return r;
}

// ---------------------------------------------------------------------------
// Kernel 1: convert fp32 weights [1024,256] -> bf16 in workspace
// ---------------------------------------------------------------------------
__global__ __launch_bounds__(256) void octk2_convert_weights(
    const float* __restrict__ w, unsigned int* __restrict__ wb, int n4) {
  int i = blockIdx.x * 256 + threadIdx.x;        // one float4 -> one uint2
  if (i < n4) {
    float4 v = reinterpret_cast<const float4*>(w)[i];
    uint2 r;
    r.x = pack_bf16x2(v.x, v.y);
    r.y = pack_bf16x2(v.z, v.w);
    reinterpret_cast<uint2*>(wb)[i] = r;
  }
}

// ---------------------------------------------------------------------------
// Kernel 2: zero exactly the empty parent rows (binary search in sorted idx)
// 32 lanes per row, each lane writes 2 float4 (row = 256 floats)
// ---------------------------------------------------------------------------
__global__ __launch_bounds__(256) void octk2_zero_empty(
    const int* __restrict__ idx, float* __restrict__ out) {
  int gid  = blockIdx.x * 256 + threadIdx.x;
  int row  = gid >> 5;                            // exact: grid covers NPARENT rows
  int lane = gid & 31;
  int lo = 0, hi = GEMM_M;
  while (lo < hi) {
    int mid = (lo + hi) >> 1;
    if (idx[mid] < row) lo = mid + 1; else hi = mid;
  }
  bool present = (lo < GEMM_M) && (idx[lo] == row);
  if (!present) {
    float4 z = make_float4(0.f, 0.f, 0.f, 0.f);
    float4* p = reinterpret_cast<float4*>(out + (size_t)row * C_OUT) + lane;
    p[0]  = z;
    p[32] = z;
  }
}

// ---------------------------------------------------------------------------
// Kernel 3: bf16 WMMA GEMM, 128x128 block tile, K-step 32, double-buffered LDS
// 8 waves: wave grid 4(M) x 2(N); each wave owns 32x64 = 2x4 WMMA tiles
// ---------------------------------------------------------------------------
__global__ __launch_bounds__(256, 1) void octk2_gemm(
    const float*          __restrict__ A,      // [GEMM_M, GEMM_K] fp32
    const unsigned short* __restrict__ Bw,     // [GEMM_K, C_OUT] bf16
    const int*            __restrict__ nempty, // [GEMM_M] sorted parent rows
    float*                __restrict__ out)    // [NPARENT, C_OUT]
{
  __shared__ __align__(16) unsigned char smem[2 * A_BUF + 2 * B_BUF];
  unsigned char* ldsA0 = smem;
  unsigned char* ldsA1 = smem + A_BUF;
  unsigned char* ldsB0 = smem + 2 * A_BUF;
  unsigned char* ldsB1 = smem + 2 * A_BUF + B_BUF;

  const int tid  = threadIdx.x;
  const int lane = tid & 31;
  const int wid  = tid >> 5;
  const int wm   = wid & 3;        // M wave group: rows wm*32..+31
  const int wn   = wid >> 2;       // N wave group: cols wn*64..+63

  const int mBase = blockIdx.y * BM;
  const int n0    = blockIdx.x * BN;

  // A staging: 2 threads per row, each loads 16 floats (4x float4)
  const int rowA  = tid >> 1;
  const int halfA = tid & 1;
  const float* gA = A + (size_t)(mBase + rowA) * GEMM_K + halfA * 16;

  // B staging: 512 uint4 slots (32 rows x 16), 2 per thread
  const int brow0 = tid >> 4;           // 0..15
  const int brow1 = brow0 + 16;         // 16..31
  const int bcp   = tid & 15;           // 8-bf16 chunk within row

  float4 a0, a1, a2, a3;
  uint4  bq0, bq1;

  auto loadG = [&](int kt) {
    const int k0 = kt * BK;
    const float4* pa = reinterpret_cast<const float4*>(gA + k0);
    a0 = pa[0]; a1 = pa[1]; a2 = pa[2]; a3 = pa[3];
    bq0 = *reinterpret_cast<const uint4*>(Bw + (size_t)(k0 + brow0) * C_OUT + n0 + bcp * 8);
    bq1 = *reinterpret_cast<const uint4*>(Bw + (size_t)(k0 + brow1) * C_OUT + n0 + bcp * 8);
  };

  auto storeL = [&](unsigned char* lA, unsigned char* lB) {
    uint4 q0, q1;
    q0.x = pack_bf16x2(a0.x, a0.y); q0.y = pack_bf16x2(a0.z, a0.w);
    q0.z = pack_bf16x2(a1.x, a1.y); q0.w = pack_bf16x2(a1.z, a1.w);
    q1.x = pack_bf16x2(a2.x, a2.y); q1.y = pack_bf16x2(a2.z, a2.w);
    q1.z = pack_bf16x2(a3.x, a3.y); q1.w = pack_bf16x2(a3.z, a3.w);
    unsigned char* pa = lA + rowA * A_STRIDE + halfA * 32;
    *reinterpret_cast<uint4*>(pa)      = q0;
    *reinterpret_cast<uint4*>(pa + 16) = q1;
    *reinterpret_cast<uint4*>(lB + brow0 * B_STRIDE + bcp * 16) = bq0;
    *reinterpret_cast<uint4*>(lB + brow1 * B_STRIDE + bcp * 16) = bq1;
  };

  v8f acc[2][4] = {};

  auto compute = [&](const unsigned char* lA, const unsigned char* lB) {
    const int mrow = lane & 15;
    const int h    = lane >> 4;
    v16bf af[2];
#pragma unroll
    for (int tm = 0; tm < 2; ++tm) {
      // 16-bit A 16x32 layout: lanes 0-15 K=0..7/16..23, lanes 16-31 K=8..15/24..31
      const unsigned char* p = lA + (wm * 32 + tm * 16 + mrow) * A_STRIDE + h * 16;
      uint4 qlo = *reinterpret_cast<const uint4*>(p);
      uint4 qhi = *reinterpret_cast<const uint4*>(p + 32);
      v8u u = {qlo.x, qlo.y, qlo.z, qlo.w, qhi.x, qhi.y, qhi.z, qhi.w};
      af[tm] = __builtin_bit_cast(v16bf, u);
    }
    v16bf bf[4];
#pragma unroll
    for (int tn = 0; tn < 4; ++tn) {
      // B 32x16 layout: lane%16 = N column, lanes 0-15 K=0..7/16..23, lanes 16-31 K=8..15/24..31
      const unsigned char* p = lB + (wn * 64 + tn * 16 + mrow) * 2;
      v16bf v;
#pragma unroll
      for (int s = 0; s < 16; ++s) {
        const int k = (s < 8) ? (8 * h + s) : (16 + 8 * h + (s - 8));
        v[s] = *reinterpret_cast<const __bf16*>(p + k * B_STRIDE);
      }
      bf[tn] = v;
    }
#pragma unroll
    for (int tm = 0; tm < 2; ++tm)
#pragma unroll
      for (int tn = 0; tn < 4; ++tn)
        acc[tm][tn] = __builtin_amdgcn_wmma_f32_16x16x32_bf16(
            false, af[tm], false, bf[tn], (short)0, acc[tm][tn], false, false);
  };

  // --- software-pipelined K loop (last iteration peeled: no in-loop guards) ---
  loadG(0);
  storeL(ldsA0, ldsB0);
  __syncthreads();

  const unsigned char* cA = ldsA0;
  const unsigned char* cB = ldsB0;
  unsigned char*       nA = ldsA1;
  unsigned char*       nB = ldsB1;

#pragma unroll 2
  for (int kt = 0; kt < NK - 1; ++kt) {
    loadG(kt + 1);                              // overlap global latency with WMMAs
    const int kp = (kt + 2 < NK) ? (kt + 2) : (NK - 1);
    __builtin_prefetch(gA + kp * BK, 0, 3);     // global_prefetch: A stream, 2 ahead
    compute(cA, cB);
    storeL(nA, nB);
    __syncthreads();
    const unsigned char* tA = cA; cA = nA; nA = (unsigned char*)tA;
    const unsigned char* tB = cB; cB = nB; nB = (unsigned char*)tB;
  }
  compute(cA, cB);                              // peeled final K-step

  // --- epilogue: scatter rows to out[nempty_idx[m]] ---
  const int nloc = lane & 15;
  const int h    = lane >> 4;
#pragma unroll
  for (int tm = 0; tm < 2; ++tm) {
#pragma unroll
    for (int r = 0; r < 8; ++r) {
      const int mloc = wm * 32 + tm * 16 + h * 8 + r;   // C/D layout: VGPR r, half h
      const int orow = nempty[mBase + mloc];
      float* po = out + (size_t)orow * C_OUT + n0 + wn * 64 + nloc;
#pragma unroll
      for (int tn = 0; tn < 4; ++tn)
        po[tn * 16] = acc[tm][tn][r];
    }
  }
}

// ---------------------------------------------------------------------------
extern "C" void kernel_launch(void* const* d_in, const int* in_sizes, int n_in,
                              void* d_out, int out_size, void* d_ws, size_t ws_size,
                              hipStream_t stream) {
  const float* data    = (const float*)d_in[0];   // [N, C_IN] fp32
  const float* weights = (const float*)d_in[1];   // [8, C_IN, C_OUT] fp32
  const int*   nempty  = (const int*)d_in[2];     // [GEMM_M] int32 sorted
  float*       out     = (float*)d_out;           // [NPARENT, C_OUT] fp32
  unsigned short* wbf  = (unsigned short*)d_ws;   // bf16 weights, 512 KB

  // 1) weights fp32 -> bf16 (262144 elems, 4 per thread)
  octk2_convert_weights<<<65536 / 256, 256, 0, stream>>>(weights, (unsigned int*)wbf, 65536);

  // 2) zero exactly the empty parent rows (32 lanes per row)
  octk2_zero_empty<<<(NPARENT * 32) / 256, 256, 0, stream>>>(nempty, out);

  // 3) WMMA GEMM + scatter
  dim3 grid(C_OUT / BN, GEMM_M / BM);   // (2, 1024)
  octk2_gemm<<<grid, 256, 0, stream>>>(data, wbf, nempty, out);
}
